// MultiViewGAT_30863634989503
// MI455X (gfx1250) — matrix operand order repaired and checked by reading, not compiled, not run on hardware
//
#include <hip/hip_runtime.h>
#include <hip/hip_bf16.h>
#include <math.h>

// ---------------------------------------------------------------------------
// MultiViewGAT for MI455X (gfx1250, wave32, WMMA)
//   - dense projections via v_wmma_f32_16x16x32_f16, f32 accumulate
//   - shapes are statically friendly: rows %16==0, inner %32==0, cols %64==0
//     -> zero bounds checks, fully vectorized LDS fragment loads (ds_load_b128)
//   - W tile staged transposed in LDS so B fragments are contiguous per lane
//   - global_prefetch_b8 on upcoming A K-slices
//   - edge softmax-aggregation via global f32 atomics (HBM-bound phase)
// ---------------------------------------------------------------------------

#define NN    50000
#define EE    500000
#define EMAXE (EE + NN)
#define NHEADS 4
#define HIDD  64
#define ENCD  64
#define NVIEW 3
#define FUSD  128

typedef __attribute__((ext_vector_type(16))) _Float16 v16h;
typedef __attribute__((ext_vector_type(8)))  _Float16 v8h;
typedef __attribute__((ext_vector_type(2)))  _Float16 v2h;
typedef __attribute__((ext_vector_type(8)))  float    v8f;

// ------------------------------ utilities ----------------------------------

__device__ __forceinline__ float atomic_max_float(float* addr, float val) {
  int* ia = (int*)addr;
  int old = __float_as_int(*addr);
  while (__int_as_float(old) < val) {
    int assumed = old;
    old = atomicCAS(ia, assumed, __float_as_int(val));
    if (old == assumed) break;
  }
  return __int_as_float(old);
}

__global__ void fill_kernel(float* __restrict__ p, float v, size_t n) {
  size_t i = (size_t)blockIdx.x * blockDim.x + threadIdx.x;
  size_t st = (size_t)gridDim.x * blockDim.x;
  for (; i < n; i += st) p[i] = v;
}

__global__ void sum_reduce_kernel(const float* __restrict__ x, float* __restrict__ out, int n) {
  __shared__ float sh[256];
  float s = 0.f;
  for (int i = blockIdx.x * blockDim.x + threadIdx.x; i < n; i += gridDim.x * blockDim.x)
    s += x[i];
  sh[threadIdx.x] = s;
  __syncthreads();
  for (int st = 128; st > 0; st >>= 1) {
    if ((int)threadIdx.x < st) sh[threadIdx.x] += sh[threadIdx.x + st];
    __syncthreads();
  }
  if (threadIdx.x == 0) atomicAdd(out, sh[0]);
}

__global__ void scale_one_kernel(float* p, float s) {
  if (blockIdx.x == 0 && threadIdx.x == 0) p[0] *= s;
}

// ------------------------------ WMMA GEMM ----------------------------------
// C[rows x COLS] = A[rows x inner] @ W[inner x COLS] + bias,  COLS = NT*64
// block = 128 threads = 4 waves; block covers 16 rows x COLS cols (grid.y==1)
// wave w owns column tiles {w, w+4, ..., w+4*(NT-1)} (16 cols each)
// REQUIRES: rows % 16 == 0, inner % 32 == 0 (true for every call site here)
template <int NT>
__global__ void __launch_bounds__(128)
gemm_wmma_f16(const float* __restrict__ A,
              const float* __restrict__ W,
              const float* __restrict__ bias,
              float* __restrict__ C,
              int rows, int inner) {
  constexpr int COLS = NT * 64;
  __shared__ __align__(16) _Float16 sA[16][32];         // A tile, row-major
  __shared__ __align__(16) _Float16 sBt[COLS][32];      // W tile, TRANSPOSED [col][k]

  const int row0 = blockIdx.x * 16;
  const int tid  = threadIdx.x;        // 0..127
  const int wave = tid >> 5;           // 0..3
  const int lane = tid & 31;
  const int half = lane >> 4;          // 0/1
  const int m    = lane & 15;

  v8f acc[NT];
#pragma unroll
  for (int t = 0; t < NT; ++t) acc[t] = (v8f){};

  for (int k0 = 0; k0 < inner; k0 += 32) {
    // prefetch next A K-slice (global_prefetch_b8); full tiles -> no row guard
    if (k0 + 32 < inner && tid < 16) {
      __builtin_prefetch(A + (size_t)(row0 + tid) * inner + (k0 + 32), 0, 1);
    }
    // ---- stage A tile 16x32 (f32 -> f16), float2-vectorized, no guards ----
#pragma unroll
    for (int j = 0; j < 2; ++j) {                 // 256 float2 pairs / 128 thr
      int p  = tid + 128 * j;                     // 0..255
      int r  = p >> 4;                            // 16 pairs per row
      int pp = p & 15;
      float2 va = *(const float2*)(A + (size_t)(row0 + r) * inner + k0 + pp * 2);
      v2h h2; h2[0] = (_Float16)va.x; h2[1] = (_Float16)va.y;
      *(v2h*)&sA[r][pp * 2] = h2;
    }
    // ---- stage W tile 32xCOLS transposed (f32 -> f16), float2 loads -------
#pragma unroll
    for (int j = 0; j < (16 * COLS) / 128; ++j) { // 16*COLS float2 pairs total
      int p  = tid + 128 * j;
      int kk = p / (COLS / 2);
      int cp = p - kk * (COLS / 2);
      float2 vb = *(const float2*)(W + (size_t)(k0 + kk) * COLS + cp * 2);
      sBt[cp * 2][kk]     = (_Float16)vb.x;
      sBt[cp * 2 + 1][kk] = (_Float16)vb.y;
    }
    __syncthreads();

    // ---- A fragment: two contiguous 16B LDS reads per lane ----------------
    // lane<16 : row m, K {0..7,16..23}; lane>=16: row m, K {8..15,24..31}
    v8h alo = *(const v8h*)&sA[m][half * 8];
    v8h ahi = *(const v8h*)&sA[m][16 + half * 8];
    v16h afrag = __builtin_shufflevector(alo, ahi, 0, 1, 2, 3, 4, 5, 6, 7,
                                         8, 9, 10, 11, 12, 13, 14, 15);
#pragma unroll
    for (int t = 0; t < NT; ++t) {
      // B fragment: lane N = m (col), K {0..7,16..23} (lane<16) / {8..15,24..31} (lane>=16)
      const int ct = ((wave + 4 * t) << 4) + m;
      v8h blo = *(const v8h*)&sBt[ct][half * 8];
      v8h bhi = *(const v8h*)&sBt[ct][16 + half * 8];
      v16h bfrag = __builtin_shufflevector(blo, bhi, 0, 1, 2, 3, 4, 5, 6, 7,
                                           8, 9, 10, 11, 12, 13, 14, 15);
      acc[t] = __builtin_amdgcn_wmma_f32_16x16x32_f16(false, afrag, false, bfrag,
                                                      (short)0, acc[t], false, false);
    }
    __syncthreads();
  }

  // C/D layout: VGPR r: lane<16 -> (M=r, N=lane); lane>=16 -> (M=8+r, N=lane-16)
#pragma unroll
  for (int t = 0; t < NT; ++t) {
    const int ccol = ((wave + 4 * t) << 4) + m;
    const float bv = bias[ccol];
#pragma unroll
    for (int r = 0; r < 8; ++r) {
      const int crow = row0 + half * 8 + r;
      C[(size_t)crow * COLS + ccol] = acc[t][r] + bv;
    }
  }
}

// ------------------------------ GATv2 edge phase ---------------------------
// thread = (edge, head). Self-loops appended logically: e in [E, E+N) => loop.
__global__ void gat_edge_logits(const float* __restrict__ xl, const float* __restrict__ xr,
                                const int* __restrict__ esrc, const int* __restrict__ edst,
                                const float* __restrict__ w, const float* __restrict__ wmean,
                                const float* __restrict__ att, const float* __restrict__ We,
                                float* __restrict__ logit, float* __restrict__ mx,
                                int E, int N, int H, int Cc) {
  int idx = blockIdx.x * blockDim.x + threadIdx.x;
  int EM = E + N;
  if (idx >= EM * H) return;
  int e = idx / H, h = idx - e * H;
  int s, d; float we;
  if (e < E) { s = esrc[e]; d = edst[e]; we = w[e]; }
  else       { s = e - E;   d = s;       we = wmean[0]; }
  const int F = H * Cc;
  const float* pl = xl + (size_t)s * F + h * Cc;
  const float* pr = xr + (size_t)d * F + h * Cc;
  const float* pe = We + h * Cc;
  const float* pa = att + h * Cc;
  float acc = 0.f;
  for (int c = 0; c < Cc; ++c) {
    float mv = pl[c] + pr[c] + we * pe[c];
    mv = (mv >= 0.f) ? mv : 0.2f * mv;      // leaky_relu(0.2)
    acc += pa[c] * mv;
  }
  logit[(size_t)e * H + h] = acc;
  atomic_max_float(&mx[(size_t)d * H + h], acc);
}

__global__ void gat_edge_exp(const int* __restrict__ edst,
                             float* __restrict__ logit,
                             const float* __restrict__ mx, float* __restrict__ den,
                             int E, int N, int H) {
  int idx = blockIdx.x * blockDim.x + threadIdx.x;
  int EM = E + N;
  if (idx >= EM * H) return;
  int e = idx / H, h = idx - e * H;
  int d = (e < E) ? edst[e] : (e - E);
  float ex = __expf(logit[idx] - mx[(size_t)d * H + h]);
  logit[idx] = ex;
  atomicAdd(&den[(size_t)d * H + h], ex);
}

__global__ void gat_edge_aggregate(const float* __restrict__ xl,
                                   const int* __restrict__ esrc, const int* __restrict__ edst,
                                   const float* __restrict__ logit, const float* __restrict__ den,
                                   float* __restrict__ out,
                                   int E, int N, int H, int Cc) {
  int idx = blockIdx.x * blockDim.x + threadIdx.x;
  int EM = E + N;
  if (idx >= EM * H) return;
  int e = idx / H, h = idx - e * H;
  int s, d;
  if (e < E) { s = esrc[e]; d = edst[e]; }
  else       { s = e - E;   d = s; }
  float alpha = logit[idx] / fmaxf(den[(size_t)d * H + h], 1e-16f);
  const int F = H * Cc;
  const float* ps = xl + (size_t)s * F + h * Cc;
  float* po = out + (size_t)d * F + h * Cc;
  for (int c = 0; c < Cc; ++c) atomicAdd(&po[c], alpha * ps[c]);
}

// ------------------------------ BatchNorm + PReLU --------------------------
__global__ void bn_stats_kernel(const float* __restrict__ x, float* __restrict__ meanvar,
                                int rows, int cols) {
  __shared__ float sh[256], sh2[256];
  const int c = blockIdx.x;
  float s = 0.f, s2 = 0.f;
  for (int r = threadIdx.x; r < rows; r += blockDim.x) {
    float v = x[(size_t)r * cols + c];
    s += v; s2 += v * v;
  }
  sh[threadIdx.x] = s; sh2[threadIdx.x] = s2;
  __syncthreads();
  for (int st = 128; st > 0; st >>= 1) {
    if ((int)threadIdx.x < st) { sh[threadIdx.x] += sh[threadIdx.x + st]; sh2[threadIdx.x] += sh2[threadIdx.x + st]; }
    __syncthreads();
  }
  if (threadIdx.x == 0) {
    float mu = sh[0] / rows;
    float var = sh2[0] / rows - mu * mu;
    meanvar[c] = mu;
    meanvar[cols + c] = var;
  }
}

// note: the reference adds a per-column constant bias before BN; it cancels in
// (x - mean), so it is omitted here (exact same result).
__global__ void bn_prelu_kernel(float* __restrict__ x, const float* __restrict__ meanvar,
                                const float* __restrict__ g, const float* __restrict__ b,
                                const float* __restrict__ prelu, int rows, int cols) {
  size_t n = (size_t)rows * cols;
  size_t i = (size_t)blockIdx.x * blockDim.x + threadIdx.x;
  size_t st = (size_t)gridDim.x * blockDim.x;
  float a = prelu[0];
  for (; i < n; i += st) {
    int c = (int)(i % cols);
    float mu = meanvar[c], var = meanvar[cols + c];
    float y = (x[i] - mu) * rsqrtf(var + 1e-5f) * g[c] + b[c];
    x[i] = (y >= 0.f) ? y : a * y;
  }
}

// hs[v] slice of HS[N x 192]; adds skip connection
__global__ void add_skip_store(const float* __restrict__ h, const float* __restrict__ skip,
                               float* __restrict__ HS, int N, int v) {
  int idx = blockIdx.x * blockDim.x + threadIdx.x;
  if (idx >= N * ENCD) return;
  int n = idx / ENCD, c = idx - n * ENCD;
  HS[(size_t)n * (NVIEW * ENCD) + v * ENCD + c] = h[idx] + skip[idx];
}

// ------------------------------ cross attention ----------------------------
// thread = (node, head); 3x3 score matrix per head, register resident
__global__ void cross_attn_kernel(const float* __restrict__ q, const float* __restrict__ k,
                                  const float* __restrict__ vv, float* __restrict__ o, int N) {
  int idx = blockIdx.x * blockDim.x + threadIdx.x;
  if (idx >= N * NHEADS) return;
  int n = idx >> 2, h = idx & 3;
  const int hd = ENCD / NHEADS;     // 16
  float sc[NVIEW][NVIEW];
#pragma unroll
  for (int a = 0; a < NVIEW; ++a) {
    const float* qp = q + ((size_t)n * NVIEW + a) * ENCD + h * hd;
#pragma unroll
    for (int bb = 0; bb < NVIEW; ++bb) {
      const float* kp = k + ((size_t)n * NVIEW + bb) * ENCD + h * hd;
      float s = 0.f;
#pragma unroll
      for (int t = 0; t < hd; ++t) s += qp[t] * kp[t];
      sc[a][bb] = s * 0.25f;        // 1/sqrt(16)
    }
  }
#pragma unroll
  for (int a = 0; a < NVIEW; ++a) {
    float mxv = fmaxf(sc[a][0], fmaxf(sc[a][1], sc[a][2]));
    float e0 = __expf(sc[a][0] - mxv), e1 = __expf(sc[a][1] - mxv), e2 = __expf(sc[a][2] - mxv);
    float inv = 1.f / (e0 + e1 + e2);
    float w0 = e0 * inv, w1 = e1 * inv, w2 = e2 * inv;
    const float* v0 = vv + ((size_t)n * NVIEW + 0) * ENCD + h * hd;
    const float* v1 = vv + ((size_t)n * NVIEW + 1) * ENCD + h * hd;
    const float* v2 = vv + ((size_t)n * NVIEW + 2) * ENCD + h * hd;
    float* op = o + ((size_t)n * NVIEW + a) * ENCD + h * hd;
#pragma unroll
    for (int t = 0; t < hd; ++t) op[t] = w0 * v0[t] + w1 * v1[t] + w2 * v2[t];
  }
}

// ------------------------------ LN / gating / head -------------------------
__global__ void layernorm_rows(float* __restrict__ x, const float* __restrict__ g,
                               const float* __restrict__ b, int rows, int D) {
  int r = blockIdx.x * blockDim.x + threadIdx.x;
  if (r >= rows) return;
  float* p = x + (size_t)r * D;
  float mu = 0.f;
  for (int d = 0; d < D; ++d) mu += p[d];
  mu /= D;
  float var = 0.f;
  for (int d = 0; d < D; ++d) { float t = p[d] - mu; var += t * t; }
  var /= D;
  float inv = rsqrtf(var + 1e-5f);
  for (int d = 0; d < D; ++d) p[d] = (p[d] - mu) * inv * g[d] + b[d];
}

__global__ void mean_views_kernel(const float* __restrict__ o, float* __restrict__ hcross, int N) {
  int idx = blockIdx.x * blockDim.x + threadIdx.x;
  if (idx >= N * ENCD) return;
  int n = idx / ENCD, c = idx - n * ENCD;
  const float* p = o + (size_t)n * (NVIEW * ENCD) + c;
  hcross[idx] = (p[0] + p[ENCD] + p[2 * ENCD]) * (1.f / 3.f);
}

__global__ void gate_mul_kernel(const float* __restrict__ HS, float* __restrict__ gbuf, size_t n) {
  size_t i = (size_t)blockIdx.x * blockDim.x + threadIdx.x;
  size_t st = (size_t)gridDim.x * blockDim.x;
  for (; i < n; i += st) {
    float s = 1.f / (1.f + __expf(-gbuf[i]));
    gbuf[i] = HS[i] * s;
  }
}

__global__ void gelu_kernel(float* __restrict__ x, size_t n) {
  size_t i = (size_t)blockIdx.x * blockDim.x + threadIdx.x;
  size_t st = (size_t)gridDim.x * blockDim.x;
  for (; i < n; i += st) {
    float v = x[i];
    x[i] = 0.5f * v * (1.f + erff(v * 0.70710678118654752f));
  }
}

__global__ void final_head_kernel(const float* __restrict__ hfuse, const float* __restrict__ hcross,
                                  const float* __restrict__ Wh, const float* __restrict__ bh,
                                  float* __restrict__ out, int N) {
  int n = blockIdx.x * blockDim.x + threadIdx.x;
  if (n >= N) return;
  float acc = bh[0];
  const float* pf = hfuse + (size_t)n * FUSD;
  for (int i = 0; i < FUSD; ++i) acc += pf[i] * Wh[i];
  const float* pc = hcross + (size_t)n * ENCD;
  for (int i = 0; i < ENCD; ++i) acc += pc[i] * Wh[FUSD + i];
  out[n] = acc;
}

// ------------------------------ host orchestration -------------------------

extern "C" void kernel_launch(void* const* d_in, const int* in_sizes, int n_in,
                              void* d_out, int out_size, void* d_ws, size_t ws_size,
                              hipStream_t stream) {
  (void)in_sizes; (void)n_in; (void)out_size; (void)ws_size;

  const int N = NN, E = EE;
  const int IN_DIMS[NVIEW] = {128, 96, 64};

  // ---- input pointers: x0,e0,w0,x1,e1,w1,x2,e2,w2 then params pytree leaves
  // params flatten (sorted dict keys): cross(10), top(8), views(3x21)
  const float* x[NVIEW]; const int* esrc[NVIEW]; const int* edst[NVIEW]; const float* wv[NVIEW];
  for (int v = 0; v < NVIEW; ++v) {
    x[v]    = (const float*)d_in[3 * v + 0];
    esrc[v] = (const int*)d_in[3 * v + 1];
    edst[v] = esrc[v] + E;
    wv[v]   = (const float*)d_in[3 * v + 2];
  }
  auto F = [&](int i) { return (const float*)d_in[i]; };
  const int P = 9;
  // cross: Wk,Wo,Wq,Wv,bk,bo,bq,bv,lnb,lng
  const float *Wk = F(P + 0), *Wo = F(P + 1), *Wq = F(P + 2), *Wvv = F(P + 3);
  const float *bk = F(P + 4), *bo = F(P + 5), *bq = F(P + 6), *bvv = F(P + 7);
  const float *lnb = F(P + 8), *lng = F(P + 9);
  // top: Wf,Wg,Wh,bf,bg,bh,lnfb,lnfg
  const float *Wf = F(P + 10), *Wg = F(P + 11), *Wh = F(P + 12);
  const float *bf = F(P + 13), *bg = F(P + 14), *bh = F(P + 15);
  const float *lnfb = F(P + 16), *lnfg = F(P + 17);
  // views: We1,We2,Wl1,Wl2,Wr1,Wr2,att1,att2,b1,b2,bl1,bl2,bn1b,bn1g,bn2b,bn2g,br1,br2,prelu,skipW,skipb
  const int VB = P + 18;
  auto VP = [&](int v, int i) { return F(VB + v * 21 + i); };

  // ---- workspace layout (floats)
  float* ws = (float*)d_ws;
  float* HS    = ws;                                // N x 192 (persists; = h_cat and stacked H)
  float* wmean = HS + (size_t)N * 192;              // 4
  float* SCR   = wmean + 4;                         // phase-unioned scratch

  // phase-1 (per view) layout inside SCR
  float* xl1  = SCR;
  float* xr1  = xl1 + (size_t)N * 256;
  float* hb   = xr1 + (size_t)N * 256;
  float* xl2  = hb  + (size_t)N * 256;
  float* xr2  = xl2 + (size_t)N * 64;
  float* ob2  = xr2 + (size_t)N * 64;
  float* lgt  = ob2 + (size_t)N * 64;               // EMAX*4
  float* mx   = lgt + (size_t)EMAXE * 4;            // N*4
  float* den  = mx  + (size_t)N * 4;                // N*4
  float* stats= den + (size_t)N * 4;                // 512
  float* skb  = stats + 512;                        // N*64

  // rows % 16 == 0 and inner % 32 == 0 hold for every call below
  auto gemm = [&](const float* A, const float* W, const float* bias, float* Cp,
                  int rows, int inner, int cols) {
    dim3 g(rows / 16, 1);
    switch (cols) {
      case 64:  gemm_wmma_f16<1><<<g, 128, 0, stream>>>(A, W, bias, Cp, rows, inner); break;
      case 128: gemm_wmma_f16<2><<<g, 128, 0, stream>>>(A, W, bias, Cp, rows, inner); break;
      case 192: gemm_wmma_f16<3><<<g, 128, 0, stream>>>(A, W, bias, Cp, rows, inner); break;
      case 256: gemm_wmma_f16<4><<<g, 128, 0, stream>>>(A, W, bias, Cp, rows, inner); break;
      default: break;
    }
  };
  auto fill = [&](float* p, float v, size_t n) {
    fill_kernel<<<2048, 256, 0, stream>>>(p, v, n);
  };

  // =================== phase 1: per-view GATv2 encoders ====================
  for (int v = 0; v < NVIEW; ++v) {
    const int d = IN_DIMS[v];
    // mean edge weight (for self-loop edge attr)
    fill(wmean, 0.f, 1);
    sum_reduce_kernel<<<512, 256, 0, stream>>>(wv[v], wmean, E);
    scale_one_kernel<<<1, 32, 0, stream>>>(wmean, 1.f / (float)E);

    // --- layer 1 (H=4, C=64) ---
    gemm(x[v], VP(v, 2), VP(v, 10), xl1, N, d, 256);   // Wl1, bl1
    gemm(x[v], VP(v, 4), VP(v, 16), xr1, N, d, 256);   // Wr1, br1
    fill(hb, 0.f, (size_t)N * 256);
    fill(mx, -3.0e38f, (size_t)N * 4);
    fill(den, 0.f, (size_t)N * 4);
    {
      int cnt = EMAXE * 4, grid = (cnt + 255) / 256;
      gat_edge_logits<<<grid, 256, 0, stream>>>(xl1, xr1, esrc[v], edst[v], wv[v], wmean,
                                                VP(v, 6), VP(v, 0), lgt, mx, E, N, 4, 64);
      gat_edge_exp<<<grid, 256, 0, stream>>>(edst[v], lgt, mx, den, E, N, 4);
      gat_edge_aggregate<<<grid, 256, 0, stream>>>(xl1, esrc[v], edst[v], lgt, den, hb,
                                                   E, N, 4, 64);
    }
    bn_stats_kernel<<<256, 256, 0, stream>>>(hb, stats, N, 256);
    bn_prelu_kernel<<<2048, 256, 0, stream>>>(hb, stats, VP(v, 13), VP(v, 12), VP(v, 18), N, 256);

    // --- layer 2 (H=1, C=64) ---
    gemm(hb, VP(v, 3), VP(v, 11), xl2, N, 256, 64);    // Wl2, bl2
    gemm(hb, VP(v, 5), VP(v, 17), xr2, N, 256, 64);    // Wr2, br2
    fill(ob2, 0.f, (size_t)N * 64);
    fill(mx, -3.0e38f, (size_t)N);
    fill(den, 0.f, (size_t)N);
    {
      int cnt = EMAXE, grid = (cnt + 255) / 256;
      gat_edge_logits<<<grid, 256, 0, stream>>>(xl2, xr2, esrc[v], edst[v], wv[v], wmean,
                                                VP(v, 7), VP(v, 1), lgt, mx, E, N, 1, 64);
      gat_edge_exp<<<grid, 256, 0, stream>>>(edst[v], lgt, mx, den, E, N, 1);
      gat_edge_aggregate<<<grid, 256, 0, stream>>>(xl2, esrc[v], edst[v], lgt, den, ob2,
                                                   E, N, 1, 64);
    }
    bn_stats_kernel<<<64, 256, 0, stream>>>(ob2, stats, N, 64);
    bn_prelu_kernel<<<2048, 256, 0, stream>>>(ob2, stats, VP(v, 15), VP(v, 14), VP(v, 18), N, 64);

    // skip connection + write view slice into HS (= h_cat / stacked H)
    gemm(x[v], VP(v, 19), VP(v, 20), skb, N, d, 64);   // skipW, skipb
    add_skip_store<<<(N * 64 + 255) / 256, 256, 0, stream>>>(ob2, skb, HS, N, v);
  }

  // =================== phase 2: cross-attention + fusion ===================
  // reuse SCR region
  float* q    = SCR;                                 // (N*3) x 64
  float* k    = q  + (size_t)N * 192;
  float* vbuf = k  + (size_t)N * 192;
  float* o4   = vbuf + (size_t)N * 192;
  float* gbuf = o4 + (size_t)N * 192;                // N x 192 (Wo out / scratch)
  float* fbuf = gbuf + (size_t)N * 192;              // N x 128
  float* hcr  = fbuf + (size_t)N * 128;              // N x 64

  const int NR = N * NVIEW;                          // 150000 rows of dim 64
  gemm(HS, Wq, bq, q, NR, 64, 64);
  gemm(HS, Wk, bk, k, NR, 64, 64);
  gemm(HS, Wvv, bvv, vbuf, NR, 64, 64);
  cross_attn_kernel<<<(N * NHEADS + 255) / 256, 256, 0, stream>>>(q, k, vbuf, o4, N);
  gemm(o4, Wo, bo, gbuf, NR, 64, 64);
  layernorm_rows<<<(NR + 255) / 256, 256, 0, stream>>>(gbuf, lng, lnb, NR, 64);
  mean_views_kernel<<<(N * 64 + 255) / 256, 256, 0, stream>>>(gbuf, hcr, N);

  // gated fusion: h_gate = HS * sigmoid(HS@Wg + bg)  (reuse q region for gate pre-act)
  float* gpre = q;
  gemm(HS, Wg, bg, gpre, N, 192, 192);
  gate_mul_kernel<<<2048, 256, 0, stream>>>(HS, gpre, (size_t)N * 192);
  gemm(gpre, Wf, bf, fbuf, N, 192, 128);
  layernorm_rows<<<(N + 255) / 256, 256, 0, stream>>>(fbuf, lnfg, lnfb, N, 128);
  gelu_kernel<<<2048, 256, 0, stream>>>(fbuf, (size_t)N * 128);

  final_head_kernel<<<(N + 255) / 256, 256, 0, stream>>>(fbuf, hcr, Wh, bh, (float*)d_out, N);
}
